// TE_84679575208352
// MI455X (gfx1250) — matrix operand order repaired and checked by reading, not compiled, not run on hardware
//
#include <hip/hip_runtime.h>

// ---------------------------------------------------------------------------
// Hierarchical type-embedding (GRU DAG) for MI455X / gfx1250.
// All GEMMs use V_WMMA_F32_16X16X32_BF16 (bf16 inputs, f32 accumulate).
// M-tile = 32 rows/block (2 row-tiles per wave) so every B fragment is
// reused twice -> halves W/U L2 traffic vs the M=16 version.
// ---------------------------------------------------------------------------

typedef unsigned short u16;
typedef __attribute__((ext_vector_type(16))) __bf16        v16bf;
typedef __attribute__((ext_vector_type(8)))  float         v8f;
typedef __attribute__((ext_vector_type(4)))  unsigned int  v4u;

union AFrag { v16bf v; unsigned int d[8]; };
union BFrag { v16bf v; v4u q[2]; };

#define CLS   8192      // C: classes per level
#define DIM   128       // D
#define PP    8         // props
#define FF    4         // funcs
#define KK    4         // params per func
#define SS    4         // supers
#define NFG   16384     // global functions
#define LVL   8         // levels
#define NTOT  (LVL*CLS) // 65536

static __device__ __forceinline__ u16 f2bf(float f) {
  unsigned int u = __float_as_uint(f);
  u += 0x7FFFu + ((u >> 16) & 1u);     // round-to-nearest-even
  return (u16)(u >> 16);
}

static __device__ __forceinline__ float sigm(float x) {
  return 1.0f / (1.0f + __expf(-x));
}

// ---------------------------------------------------------------------------
// K=128 GEMM pass, MT row-tiles x NT col-tiles per wave:
//   acc[mt*NT+j] += A_mt(16x128 bf16) * B(128 x NT*16 bf16, stride ldb)
// A fragment layout (ISA 7.12.2, 16-bit A 16x32): lane m=lane&15, half=lane>>4;
//   dword v holds K pair { kbase, kbase+1 }, kbase = v<4 ? 2v+8*half
//                                                       : 16+2(v-4)+8*half.
// B fragment layout: lane N = lane&15, dword v holds the same K pair as A
//   (gathered u16 loads). Each B fragment feeds MT WMMAs.
// ---------------------------------------------------------------------------
template<int NT, int MT>
static __device__ __forceinline__ void gemm_k128(v8f* acc, const u16* const* aRow,
                                                 const u16* B, int ldb, int lane) {
  const int half = lane >> 4;
  #pragma unroll
  for (int kt = 0; kt < 4; ++kt) {
    const int k0 = kt * 32;
    AFrag a[MT];
    #pragma unroll
    for (int mt = 0; mt < MT; ++mt)
      #pragma unroll
      for (int v = 0; v < 8; ++v) {
        const int kk = k0 + ((v < 4) ? (2*v + 8*half) : (16 + 2*(v-4) + 8*half));
        a[mt].d[v] = *reinterpret_cast<const unsigned int*>(aRow[mt] + kk);
      }
    #pragma unroll
    for (int j = 0; j < NT; ++j) {
      BFrag b;
      const u16* Bc = B + j*16 + (lane & 15);
      #pragma unroll
      for (int v = 0; v < 8; ++v) {
        const int kk = k0 + ((v < 4) ? (2*v + 8*half) : (16 + 2*(v-4) + 8*half));
        b.q[v >> 2][v & 3] = (unsigned int)Bc[(size_t)kk * ldb] |
                             ((unsigned int)Bc[(size_t)(kk + 1) * ldb] << 16);
      }
      #pragma unroll
      for (int mt = 0; mt < MT; ++mt)
        acc[mt*NT + j] = __builtin_amdgcn_wmma_f32_16x16x32_bf16(
            false, a[mt].v, false, b.v, (short)0, acc[mt*NT + j], false, false);
    }
  }
}

// C/D layout (ISA 7.12.2, 32-bit C 16x16): vgpr jj -> M = jj + 8*half, N = lane&15
template<int NT, int MT>
static __device__ __forceinline__ void write_tiles_lds(float* lds, const v8f* acc,
                                                       int colBase, int lane) {
  const int half = lane >> 4, m = lane & 15;
  #pragma unroll
  for (int mt = 0; mt < MT; ++mt)
    #pragma unroll
    for (int j = 0; j < NT; ++j)
      #pragma unroll
      for (int jj = 0; jj < 8; ++jj)
        lds[(mt*16 + jj + 8*half) * 512 + colBase + j*16 + m] = acc[mt*NT + j][jj];
}

// ---------------------------------------------------------------------------
// Fused GRU step (Keras reset_after):
//   cols 0..255   : s_zr = x@W[:,0:256] + h@U[:,0:256]   (waves 0..3)
//   cols 256..383 : xn   = x@W[:,256:384]                (waves 4..5)
//   cols 384..511 : hn   = h@U[:,256:384]                (waves 6..7)
// then h' = z*h + (1-z)*tanh(xn+bi_n + r*(hn+br_n)) elementwise via LDS.
// x row source: gathered (xidx) / strided (xrowStride,xoff) / broadcast.
// Block = 256 threads = 8 waves, 32 rows per block. Rows are block-private.
// ---------------------------------------------------------------------------
__global__ __launch_bounds__(256) void gru_step_kernel(
    const u16* __restrict__ xbase, const int* __restrict__ xidx, int idxStride,
    int xrowStride, int xoff,
    const u16* __restrict__ W, const u16* __restrict__ U,
    const float* __restrict__ bi, const float* __restrict__ br,
    float* __restrict__ hF, u16* __restrict__ hB) {
  __shared__ float lds[32 * 512];
  const int lane = threadIdx.x & 31;
  const int w    = threadIdx.x >> 5;
  const int m    = lane & 15;
  const int r0   = blockIdx.x * 32;

  const u16* xrow[2];
  const u16* hrow[2];
  #pragma unroll
  for (int mt = 0; mt < 2; ++mt) {
    const int row = r0 + mt*16 + m;
    xrow[mt] = xidx
        ? xbase + (size_t)xidx[(size_t)row * idxStride] * DIM
        : xbase + (size_t)row * xrowStride + xoff;
    hrow[mt] = hB + (size_t)row * DIM;
  }

  v8f acc[8];                                // [mt][j]
  #pragma unroll
  for (int j = 0; j < 8; ++j)
    #pragma unroll
    for (int e = 0; e < 8; ++e) acc[j][e] = 0.0f;

  if (w < 4) {                               // s_zr = x@Wzr + h@Uzr
    const int bcol = w * 64;
    gemm_k128<4,2>(acc, xrow, W + bcol, 384, lane);
    gemm_k128<4,2>(acc, hrow, U + bcol, 384, lane);
    write_tiles_lds<4,2>(lds, acc, bcol, lane);
  } else if (w < 6) {                        // xn
    const int bcol = 256 + (w - 4) * 64;
    gemm_k128<4,2>(acc, xrow, W + bcol, 384, lane);
    write_tiles_lds<4,2>(lds, acc, bcol, lane);
  } else {                                   // hn  -> lds cols 384..511
    const int bcol = 256 + (w - 6) * 64;
    gemm_k128<4,2>(acc, hrow, U + bcol, 384, lane);
    write_tiles_lds<4,2>(lds, acc, bcol + 128, lane);
  }
  __syncthreads();

  #pragma unroll
  for (int i = 0; i < 16; ++i) {
    const int lin = threadIdx.x + i * 256;   // 0..4095
    const int rr  = lin >> 7;                // 0..31
    const int d   = lin & 127;
    const float z  = sigm(lds[rr*512 + d]       + bi[d]       + br[d]);
    const float r  = sigm(lds[rr*512 + 128 + d] + bi[128 + d] + br[128 + d]);
    const float xn = lds[rr*512 + 256 + d] + bi[256 + d];
    const float hn = lds[rr*512 + 384 + d] + br[256 + d];
    const size_t gi = (size_t)(r0 + rr) * DIM + d;
    const float hprev = hF[gi];
    const float nn = tanhf(xn + r * hn);
    const float hnew = z * hprev + (1.0f - z) * nn;
    hF[gi] = hnew;
    hB[gi] = f2bf(hnew);
  }
}

// ---------------------------------------------------------------------------
// out = [h | table[retIdx]] @ W_fn   (K=256, Ncols=128).
// 8 waves x 1 col-tile x 2 row-tiles; 32 rows per block.
// ---------------------------------------------------------------------------
__global__ __launch_bounds__(256) void fc_kernel(
    const u16* __restrict__ hB, const u16* __restrict__ tableBf,
    const int* __restrict__ retIdx, const u16* __restrict__ Wfn,
    float* __restrict__ outF, u16* __restrict__ outB) {
  const int lane = threadIdx.x & 31;
  const int w    = threadIdx.x >> 5;
  const int half = lane >> 4, m = lane & 15;
  const int r0   = blockIdx.x * 32;

  const u16* hrow[2];
  const u16* rrow[2];
  #pragma unroll
  for (int mt = 0; mt < 2; ++mt) {
    const int row = r0 + mt*16 + m;
    hrow[mt] = hB + (size_t)row * DIM;
    rrow[mt] = tableBf + (size_t)retIdx[row] * DIM;
  }

  v8f acc[2];
  #pragma unroll
  for (int j = 0; j < 2; ++j)
    #pragma unroll
    for (int e = 0; e < 8; ++e) acc[j][e] = 0.0f;

  const int bcol = w * 16;
  gemm_k128<1,2>(acc, hrow, Wfn + bcol, DIM, lane);                     // W_fn rows 0..127
  gemm_k128<1,2>(acc, rrow, Wfn + (size_t)128 * DIM + bcol, DIM, lane); // rows 128..255

  #pragma unroll
  for (int mt = 0; mt < 2; ++mt)
    #pragma unroll
    for (int jj = 0; jj < 8; ++jj) {
      const int orow = r0 + mt*16 + jj + 8 * half;
      const int oc   = bcol + m;
      const float vv = acc[mt][jj];
      outF[(size_t)orow * DIM + oc] = vv;
      if (outB) outB[(size_t)orow * DIM + oc] = f2bf(vv);
    }
}

// -------------------------- small utility kernels --------------------------
__global__ void cvt_bf16_kernel(const float* __restrict__ src,
                                u16* __restrict__ dst, int n) {
  int i = blockIdx.x * 256 + threadIdx.x;
  if (i < n) dst[i] = f2bf(src[i]);
}

__global__ void zero_h_kernel(float* __restrict__ hF, u16* __restrict__ hB, int n) {
  int i = blockIdx.x * 256 + threadIdx.x;
  if (i < n) { hF[i] = 0.0f; hB[i] = 0; }
}

__global__ void init_table0_kernel(const float* __restrict__ basic_emb,
                                   const int* __restrict__ basic_ids,
                                   float* __restrict__ table,
                                   u16* __restrict__ tableBf) {
  int i = blockIdx.x * 256 + threadIdx.x;          // over CLS*DIM
  int row = i >> 7, d = i & 127;
  float v = basic_emb[basic_ids[row] * DIM + d];
  table[i] = v;
  tableBf[i] = f2bf(v);
}

__global__ void store_rows_kernel(const float* __restrict__ h,
                                  float* __restrict__ dstF,
                                  u16* __restrict__ dstB, int n) {
  int i = blockIdx.x * 256 + threadIdx.x;
  if (i < n) { float v = h[i]; dstF[i] = v; dstB[i] = f2bf(v); }
}

// ---------------------------------------------------------------------------
extern "C" void kernel_launch(void* const* d_in, const int* in_sizes, int n_in,
                              void* d_out, int out_size, void* d_ws, size_t ws_size,
                              hipStream_t stream) {
  (void)in_sizes; (void)n_in; (void)out_size; (void)ws_size;

  const int*   basic_ids = (const int*)  d_in[0];
  const int*   prop_idx  = (const int*)  d_in[1];   // [7,C,P]
  const int*   fpi       = (const int*)  d_in[2];   // [7,C,F,K]
  const int*   fri       = (const int*)  d_in[3];   // [7,C,F]
  const int*   sup       = (const int*)  d_in[4];   // [7,C,S]
  const int*   gpi       = (const int*)  d_in[5];   // [NF,K]
  const int*   gri       = (const int*)  d_in[6];   // [NF]
  const float* basic_emb = (const float*)d_in[7];
  const float* empty_p   = (const float*)d_in[8];
  const float* empty_m   = (const float*)d_in[9];
  const float* W_fn      = (const float*)d_in[10];  // [256,128]
  const float* pW        = (const float*)d_in[11];
  const float* pU        = (const float*)d_in[12];
  const float* pb        = (const float*)d_in[13];  // [2,384]
  const float* mW        = (const float*)d_in[14];
  const float* mU        = (const float*)d_in[15];
  const float* mb        = (const float*)d_in[16];
  const float* sW        = (const float*)d_in[17];
  const float* sU        = (const float*)d_in[18];
  const float* sb        = (const float*)d_in[19];

  float* table = (float*)d_out;                        // [N,128]
  float* glob  = table + (size_t)NTOT * DIM;           // [NF,128]
  float* memfn = glob + (size_t)NFG * DIM;             // [7,C,F,128]

  // ---- workspace carve (bf16 buffers as u16) ----
  char* wp = (char*)d_ws;
  auto carve = [&](size_t bytes) -> void* {
    void* r = (void*)wp; wp += (bytes + 255) & ~(size_t)255; return r;
  };
  u16* pWb  = (u16*)carve((size_t)128 * 384 * 2);
  u16* pUb  = (u16*)carve((size_t)128 * 384 * 2);
  u16* mWb  = (u16*)carve((size_t)128 * 384 * 2);
  u16* mUb  = (u16*)carve((size_t)128 * 384 * 2);
  u16* sWb  = (u16*)carve((size_t)128 * 384 * 2);
  u16* sUb  = (u16*)carve((size_t)128 * 384 * 2);
  u16* wfnb = (u16*)carve((size_t)256 * 128 * 2);
  u16* epb  = (u16*)carve((size_t)DIM * 2);
  u16* emb  = (u16*)carve((size_t)DIM * 2);
  u16* tableBf = (u16*)carve((size_t)NTOT * DIM * 2);
  float* hP  = (float*)carve((size_t)CLS * FF * DIM * 4);   // 32768 rows
  u16*   hPb = (u16*)  carve((size_t)CLS * FF * DIM * 2);
  float* hM  = (float*)carve((size_t)CLS * DIM * 4);
  u16*   hMb = (u16*)  carve((size_t)CLS * DIM * 2);
  float* hS  = (float*)carve((size_t)CLS * DIM * 4);
  u16*   hSb = (u16*)  carve((size_t)CLS * DIM * 2);
  u16*   fnBf = (u16*) carve((size_t)CLS * FF * DIM * 2);

  const float *pbi = pb, *pbr = pb + 384;
  const float *mbi = mb, *mbr = mb + 384;
  const float *sbi = sb, *sbr = sb + 384;

  auto cvt = [&](const float* s, u16* d, int n) {
    cvt_bf16_kernel<<<(n + 255) / 256, 256, 0, stream>>>(s, d, n);
  };

  // ---- weights & constants to bf16 ----
  cvt(pW, pWb, 128 * 384); cvt(pU, pUb, 128 * 384);
  cvt(mW, mWb, 128 * 384); cvt(mU, mUb, 128 * 384);
  cvt(sW, sWb, 128 * 384); cvt(sU, sUb, 128 * 384);
  cvt(W_fn, wfnb, 256 * 128);
  cvt(empty_p, epb, DIM);  cvt(empty_m, emb, DIM);

  // ---- level 0: table[:C] = basic_emb[basic_ids] ----
  init_table0_kernel<<<(CLS * DIM) / 256, 256, 0, stream>>>(
      basic_emb, basic_ids, table, tableBf);

  const int nP = CLS * FF;              // 32768 fn rows per level
  auto gru = [&](int n, const u16* xbase, const int* xidx, int idxStride,
                 int xrowStride, int xoff, const u16* Wb, const u16* Ub,
                 const float* bi, const float* br, float* hF, u16* hB) {
    gru_step_kernel<<<n / 32, 256, 0, stream>>>(
        xbase, xidx, idxStride, xrowStride, xoff, Wb, Ub, bi, br, hF, hB);
  };

  for (int l = 1; l < LVL; ++l) {
    const int* fpi_l  = fpi  + (size_t)(l - 1) * CLS * FF * KK;
    const int* fri_l  = fri  + (size_t)(l - 1) * CLS * FF;
    const int* prop_l = prop_idx + (size_t)(l - 1) * CLS * PP;
    const int* sup_l  = sup  + (size_t)(l - 1) * CLS * SS;

    // ---- pGRU over [empty_params, 4 gathered param embeddings] ----
    zero_h_kernel<<<(nP * DIM) / 256, 256, 0, stream>>>(hP, hPb, nP * DIM);
    gru(nP, epb, nullptr, 0, 0, 0, pWb, pUb, pbi, pbr, hP, hPb);        // t=0
    for (int t = 1; t < 5; ++t)
      gru(nP, tableBf, fpi_l + (t - 1), KK, 0, 0, pWb, pUb, pbi, pbr, hP, hPb);
    // fn_e = [h|ret] @ W_fn  -> member_fn slice + bf16 mirror
    fc_kernel<<<nP / 32, 256, 0, stream>>>(
        hPb, tableBf, fri_l, wfnb, memfn + (size_t)(l - 1) * nP * DIM, fnBf);

    // ---- member GRU over [empty_members, 8 props, 4 fn_e] ----
    zero_h_kernel<<<(CLS * DIM) / 256, 256, 0, stream>>>(hM, hMb, CLS * DIM);
    gru(CLS, emb, nullptr, 0, 0, 0, mWb, mUb, mbi, mbr, hM, hMb);       // t=0
    for (int t = 1; t < 9; ++t)
      gru(CLS, tableBf, prop_l + (t - 1), PP, 0, 0, mWb, mUb, mbi, mbr, hM, hMb);
    for (int t = 9; t < 13; ++t)
      gru(CLS, fnBf, nullptr, 0, FF * DIM, (t - 9) * DIM, mWb, mUb, mbi, mbr, hM, hMb);

    // ---- super GRU over [4 gathered supers, m] ----
    zero_h_kernel<<<(CLS * DIM) / 256, 256, 0, stream>>>(hS, hSb, CLS * DIM);
    for (int t = 0; t < 4; ++t)
      gru(CLS, tableBf, sup_l + t, SS, 0, 0, sWb, sUb, sbi, sbr, hS, hSb);
    gru(CLS, hMb, nullptr, 0, DIM, 0, sWb, sUb, sbi, sbr, hS, hSb);     // t=4: m

    // table[l*C:(l+1)*C] = e  (f32 + bf16 mirror)
    store_rows_kernel<<<(CLS * DIM) / 256, 256, 0, stream>>>(
        hS, table + (size_t)l * CLS * DIM, tableBf + (size_t)l * CLS * DIM,
        CLS * DIM);
  }

  // ---- global functions ----
  zero_h_kernel<<<(NFG * DIM) / 256, 256, 0, stream>>>(hP, hPb, NFG * DIM);
  gru(NFG, epb, nullptr, 0, 0, 0, pWb, pUb, pbi, pbr, hP, hPb);         // t=0
  for (int t = 1; t < 5; ++t)
    gru(NFG, tableBf, gpi + (t - 1), KK, 0, 0, pWb, pUb, pbi, pbr, hP, hPb);
  fc_kernel<<<NFG / 32, 256, 0, stream>>>(hPb, tableBf, gri, wfnb, glob, nullptr);
}
